// Head_86131274154813
// MI455X (gfx1250) — matrix operand order repaired and checked by reading, not compiled, not run on hardware
//
#include <hip/hip_runtime.h>

typedef __attribute__((ext_vector_type(16))) __bf16 v16bf;
typedef __attribute__((ext_vector_type(8)))  __bf16 bfx8;
typedef __attribute__((ext_vector_type(8)))  float  v8f;

#define HS      64
#define TLEN    4096
#define BATCH   4
#define CDIM    1024
#define WIN     512
#define NROWS   (BATCH*TLEN)          // 16384 total rows
#define KSC     0.18033688011112042f  // log2(e)/8  (softmax scale 1/sqrt(64))
#define NEGBIG  (-3.0e30f)

#if __has_builtin(__builtin_amdgcn_global_load_async_to_lds_b128)
#define HAVE_ASYNC 1
typedef int v4i_ __attribute__((vector_size(16)));
typedef __attribute__((address_space(1))) v4i_* gasv4_p;   // AS1 (printed __device__)
typedef __attribute__((address_space(3))) v4i_* lasv4_p;   // AS3 (printed __shared__)
#else
#define HAVE_ASYNC 0
#endif

static __device__ __forceinline__ v16bf join8(bfx8 a, bfx8 b) {
    return __builtin_shufflevector(a, b, 0,1,2,3,4,5,6,7,8,9,10,11,12,13,14,15);
}
static __device__ __forceinline__ v8f wmma_bf16(v16bf A, v16bf B, v8f C) {
    return __builtin_amdgcn_wmma_f32_16x16x32_bf16(false, A, false, B, (short)0, C,
                                                   false, false);
}

// ---------------------------------------------------------------------------
// Kernel 0: convert the three [1024 x 64] fp32 weights into bf16 B-fragment
// order. Fragment for (weight w, col-tile ct, k-chunk kc): lane L owns column
// N = ct*16 + (L&15) and K-values kc*32 + (L&16) + {0..15}, stored contiguous
// (32 B per lane) so qkv_proj reads it with two b128 loads.
// ---------------------------------------------------------------------------
__global__ __launch_bounds__(128) void prep_w(const float* __restrict__ wq,
                                              const float* __restrict__ wk,
                                              const float* __restrict__ wv,
                                              __bf16* __restrict__ wfrag) {
    int t = blockIdx.x * 128 + threadIdx.x;       // 12288 threads total
    int widx = t >> 12;                           // 0=q 1=k 2=v
    int rem  = t & 4095;
    int ct   = rem >> 10;
    int kc   = (rem >> 5) & 31;
    int lane = rem & 31;
    const float* w = (widx == 0) ? wq : (widx == 1) ? wk : wv;
    int col = ct * 16 + (lane & 15);
    int k0  = kc * 32 + (lane & 16);
    alignas(16) __bf16 vals[16];
#pragma unroll
    for (int i = 0; i < 16; ++i)
        vals[i] = (__bf16)w[(size_t)(k0 + i) * HS + col];
    bfx8* dst = (bfx8*)(wfrag + (size_t)t * 16);
    dst[0] = ((bfx8*)vals)[0];
    dst[1] = ((bfx8*)vals)[1];
}

// ---------------------------------------------------------------------------
// Kernel 1: q/k/v projection. One 16-row x tile per block, staged once into
// LDS as bf16 (stride 1040 shorts -> conflict-free b128 A-fragment reads).
// 4 waves = 4 column tiles of 16; each wave keeps 3 f32 accumulators and does
// 32 K-steps x 3 weights = 96 wmma_f32_16x16x32_bf16.
// q,k stored row-major [row][64]; v stored transposed VT[b][d][t].
// ---------------------------------------------------------------------------
__global__ __launch_bounds__(128) void qkv_proj(const float* __restrict__ x,
                                                const __bf16* __restrict__ wfrag,
                                                __bf16* __restrict__ Q,
                                                __bf16* __restrict__ Kst,
                                                __bf16* __restrict__ VT) {
    constexpr int ROWSTR = 1040;                  // shorts; 2080 B, 16B aligned
    __shared__ __bf16 lds[16 * ROWSTR];

    const int tid = threadIdx.x;
    const size_t rowbase = (size_t)blockIdx.x * 16;

    // Stage: 16 rows x 1024 f32 -> bf16 LDS (x read exactly once from HBM)
    for (int g = tid; g < 16 * 256; g += 128) {
        int r  = g >> 8;
        int c4 = g & 255;
        const float4 xv = ((const float4*)(x + (rowbase + r) * CDIM))[c4];
        __bf16* p = &lds[r * ROWSTR + c4 * 4];
        p[0] = (__bf16)xv.x; p[1] = (__bf16)xv.y;
        p[2] = (__bf16)xv.z; p[3] = (__bf16)xv.w;
    }
    __syncthreads();

    const int wave = tid >> 5;                    // = column tile (0..3)
    const int lane = tid & 31;
    const int mrow = lane & 15;
    const int hi   = lane & 16;
    const __bf16* arow = &lds[mrow * ROWSTR];
    const int adelta = hi ? 16 : 0;

    v8f acc[3] = {v8f{}, v8f{}, v8f{}};

    for (int kc = 0; kc < 32; ++kc) {
        const int k = kc * 32;
        // A fragment (shared by the 3 weights): contiguous k = 16h + s order, as packed in wfrag
        bfx8 alo = *(const bfx8*)(arow + k + adelta);
        bfx8 ahi = *(const bfx8*)(arow + k + adelta + 8);
        v16bf A = join8(alo, ahi);
#pragma unroll
        for (int w = 0; w < 3; ++w) {
            const __bf16* bp =
                wfrag + ((((size_t)w * 4 + wave) * 32 + kc) * 32 + lane) * 16;
            v16bf B = join8(((const bfx8*)bp)[0], ((const bfx8*)bp)[1]);
            acc[w] = wmma_bf16(A, B, acc[w]);
        }
    }

    const int col   = wave * 16 + mrow;
    const int rbase = hi ? 8 : 0;
    // q, k row-major
#pragma unroll
    for (int r = 0; r < 8; ++r) {
        size_t row = rowbase + rbase + r;
        Q  [row * HS + col] = (__bf16)acc[0][r];
        Kst[row * HS + col] = (__bf16)acc[1][r];
    }
    // v transposed: VT[b][d][t], 8 contiguous t's -> one b128 store
    {
        size_t grow = rowbase + rbase;
        size_t b    = grow >> 12;
        size_t tpos = grow & (TLEN - 1);
        alignas(16) __bf16 vv[8];
#pragma unroll
        for (int r = 0; r < 8; ++r) vv[r] = (__bf16)acc[2][r];
        *(bfx8*)(VT + (b * HS + col) * TLEN + tpos) = *(bfx8*)vv;
    }
}

// ---------------------------------------------------------------------------
// Kernel 2: sliding-window flash attention. One 16-row q tile per wave,
// 32-key chunks: 4 score WMMAs + 4 PV WMMAs per chunk. Online softmax with
// per-row max/sum reduced across the 16 lanes holding each row (shfl_xor
// 1/2/4/8 stays inside each wave32 half). P transposed D->A layout via a
// small per-wave LDS tile.
//
// K tiles (32 rows x 128 B, contiguous in the row-major K buffer) are
// double-buffered into LDS with GLOBAL_LOAD_ASYNC_TO_LDS_B128: the next
// chunk's tile is issued before computing the current one, and we only wait
// ASYNCcnt<=8 so the prefetch overlaps with the WMMAs. LDS rows get a 144 B
// pitch (per-lane async LDS addresses make padding free) so the B-fragment
// ds_load_b128 reads avoid bank conflicts.
// ---------------------------------------------------------------------------
#define KROWPAD 72   // shorts per LDS K row (144 B pitch)

__global__ __launch_bounds__(128) void attn(const __bf16* __restrict__ Q,
                                            const __bf16* __restrict__ Kst,
                                            const __bf16* __restrict__ VT,
                                            float* __restrict__ out) {
    __shared__ __bf16 plds[4][16 * 40];           // per-wave 16x32 P tile, padded
#if HAVE_ASYNC
    __shared__ __bf16 kbuf[4][2][32 * KROWPAD];   // per-wave double-buffered K tile
#endif

    const int wave = threadIdx.x >> 5;
    const int lane = threadIdx.x & 31;
    const int qt   = blockIdx.x * 4 + wave;       // 0..1023
    const int t0g  = qt * 16;                     // global row
    const int b    = t0g >> 12;
    const int t0   = t0g & (TLEN - 1);            // row within batch

    const __bf16* Qb = Q + (size_t)t0g * HS;
    const __bf16* Kb = Kst + (size_t)b * TLEN * HS;
    const __bf16* Vb = VT + (size_t)b * HS * TLEN;

    const int mrow = lane & 15;
    const int hi   = lane & 16;
    const int ad   = hi ? 16 : 0;                 // K-delta of upper lane half (contiguous k order)
    __bf16* pb = &plds[wave][0];

    // q A-fragments for d=[0,32) and [32,64)
    const __bf16* qrow = Qb + mrow * HS;
    v16bf qa0 = join8(*(const bfx8*)(qrow + ad),      *(const bfx8*)(qrow + ad + 8));
    v16bf qa1 = join8(*(const bfx8*)(qrow + 32 + ad), *(const bfx8*)(qrow + 32 + ad + 8));

    v8f o[4] = {v8f{}, v8f{}, v8f{}, v8f{}};
    float m[8], l[8];
#pragma unroll
    for (int r = 0; r < 8; ++r) { m[r] = NEGBIG; l[r] = 0.0f; }

    const int lo  = t0 - (WIN - 16);              // lowest possibly-valid key
    const int cs0 = (lo <= 0) ? 0 : (lo & ~31);

#if HAVE_ASYNC
    __bf16* kb[2] = { &kbuf[wave][0][0], &kbuf[wave][1][0] };
    const int arow = lane >> 3;                   // 0..3 rows per issue
    const int achk = (lane & 7) * 16;             // 16B chunk within 128B row
    // preload first K tile into buffer 0
    {
        const char* g0 = (const char*)(Kb + (size_t)cs0 * HS);
        char*       l0 = (char*)kb[0];
#pragma unroll
        for (int i = 0; i < 8; ++i)
            __builtin_amdgcn_global_load_async_to_lds_b128(
                (gasv4_p)(g0 + (i * 4 + arow) * 128 + achk),
                (lasv4_p)(l0 + (i * 4 + arow) * 144 + achk), 0, 0);
    }
    int buf = 0;
#endif

    for (int cs = cs0; cs <= t0; cs += 32) {
#if HAVE_ASYNC
        const bool pre = (cs + 32 <= t0);
        if (pre) {                                // prefetch next tile -> other buf
            const char* gn = (const char*)(Kb + (size_t)(cs + 32) * HS);
            char*       ln = (char*)kb[buf ^ 1];
#pragma unroll
            for (int i = 0; i < 8; ++i)
                __builtin_amdgcn_global_load_async_to_lds_b128(
                    (gasv4_p)(gn + (i * 4 + arow) * 128 + achk),
                    (lasv4_p)(ln + (i * 4 + arow) * 144 + achk), 0, 0);
            asm volatile("s_wait_asynccnt 0x8" ::: "memory");   // drain current buf
        } else {
            asm volatile("s_wait_asynccnt 0x0" ::: "memory");
        }
#endif
        // ---- scores: S[16x32] = q @ k^T, two 16x16 D tiles -------------
        v8f sc0 = {}, sc1 = {};
        {
            const int dk = hi ? 16 : 0;
#if HAVE_ASYNC
            const __bf16* kr0 = kb[buf] + (size_t)mrow * KROWPAD;
            const __bf16* kr1 = kr0 + 16 * KROWPAD;
#else
            const __bf16* kr0 = Kb + (size_t)(cs + mrow) * HS;
            const __bf16* kr1 = kr0 + 16 * HS;
            __builtin_prefetch(kr0 + 32 * HS, 0, 1);   // next chunk
#endif
            sc0 = wmma_bf16(qa0, join8(*(const bfx8*)(kr0 + dk),
                                       *(const bfx8*)(kr0 + dk + 8)), sc0);
            sc0 = wmma_bf16(qa1, join8(*(const bfx8*)(kr0 + 32 + dk),
                                       *(const bfx8*)(kr0 + 32 + dk + 8)), sc0);
            sc1 = wmma_bf16(qa0, join8(*(const bfx8*)(kr1 + dk),
                                       *(const bfx8*)(kr1 + dk + 8)), sc1);
            sc1 = wmma_bf16(qa1, join8(*(const bfx8*)(kr1 + 32 + dk),
                                       *(const bfx8*)(kr1 + 32 + dk + 8)), sc1);
        }

        // ---- mask (edge chunks only) -----------------------------------
        const bool full = (cs + 31 <= t0) && (cs >= t0 - (WIN - 16));
        if (!full) {
            const int j0 = cs + mrow;
#pragma unroll
            for (int r = 0; r < 8; ++r) {
                int i = t0 + (hi ? 8 : 0) + r;
                int d0 = i - j0;                  // tile jt=0
                int d1 = d0 - 16;                 // tile jt=1
                if (d0 < 0 || d0 >= WIN) sc0[r] = NEGBIG;
                if (d1 < 0 || d1 >= WIN) sc1[r] = NEGBIG;
            }
        }

        // ---- online softmax --------------------------------------------
#pragma unroll
        for (int r = 0; r < 8; ++r) {
            float tm = fmaxf(sc0[r], sc1[r]);
            tm = fmaxf(tm, __shfl_xor(tm, 1));
            tm = fmaxf(tm, __shfl_xor(tm, 2));
            tm = fmaxf(tm, __shfl_xor(tm, 4));
            tm = fmaxf(tm, __shfl_xor(tm, 8));
            float mn   = fmaxf(m[r], tm);
            float corr = exp2f((m[r] - mn) * KSC);
            m[r] = mn;
            float p0 = exp2f((sc0[r] - mn) * KSC);
            float p1 = exp2f((sc1[r] - mn) * KSC);
            float rs = p0 + p1;
            rs += __shfl_xor(rs, 1);
            rs += __shfl_xor(rs, 2);
            rs += __shfl_xor(rs, 4);
            rs += __shfl_xor(rs, 8);
            l[r] = l[r] * corr + rs;
            o[0][r] *= corr; o[1][r] *= corr; o[2][r] *= corr; o[3][r] *= corr;
            int row = (hi ? 8 : 0) + r;
            pb[row * 40 + mrow]      = (__bf16)p0;   // D->A transpose via LDS
            pb[row * 40 + 16 + mrow] = (__bf16)p1;
        }

        // ---- P (A layout) and out += P @ V ------------------------------
        v16bf Ap = join8(*(const bfx8*)(pb + mrow * 40 + ad),
                         *(const bfx8*)(pb + mrow * 40 + ad + 8));
        const int sv = cs + (hi ? 16 : 0);
#pragma unroll
        for (int ct = 0; ct < 4; ++ct) {
            const __bf16* vr = Vb + (size_t)(ct * 16 + mrow) * TLEN + sv;
            v16bf Bv = join8(*(const bfx8*)vr, *(const bfx8*)(vr + 8));
            o[ct] = wmma_bf16(Ap, Bv, o[ct]);
        }
#if HAVE_ASYNC
        buf ^= 1;
#endif
    }

    // ---- normalize + store f32 output ----------------------------------
#pragma unroll
    for (int r = 0; r < 8; ++r) {
        float inv = 1.0f / l[r];
        size_t row = (size_t)t0g + (hi ? 8 : 0) + r;
        float* op = out + row * HS + mrow;
        op[0]  = o[0][r] * inv;
        op[16] = o[1][r] * inv;
        op[32] = o[2][r] * inv;
        op[48] = o[3][r] * inv;
    }
}

// ---------------------------------------------------------------------------
extern "C" void kernel_launch(void* const* d_in, const int* in_sizes, int n_in,
                              void* d_out, int out_size, void* d_ws, size_t ws_size,
                              hipStream_t stream) {
    const float* x  = (const float*)d_in[0];
    const float* wk = (const float*)d_in[1];
    const float* wq = (const float*)d_in[2];
    const float* wv = (const float*)d_in[3];
    float* out = (float*)d_out;

    // workspace layout (bf16):
    //   wfrag: 3*4*32*512   = 196608 elems (384 KB)
    //   Q,K  : 16384*64 each
    //   VT   : [4][64][4096]
    __bf16* wfrag = (__bf16*)d_ws;
    __bf16* Qs    = wfrag + 196608;
    __bf16* Ks    = Qs + (size_t)NROWS * HS;
    __bf16* VTs   = Ks + (size_t)NROWS * HS;

    prep_w  <<<96,   128, 0, stream>>>(wq, wk, wv, wfrag);
    qkv_proj<<<1024, 128, 0, stream>>>(x, wfrag, Qs, Ks, VTs);
    attn    <<<256,  128, 0, stream>>>(Qs, Ks, VTs, out);
}
